// SeqMamba_Module_74990128988639
// MI455X (gfx1250) — compile-verified, matrix operand order checked by reading, not run on hardware
//
#include <hip/hip_runtime.h>
#include <hip/hip_bf16.h>
#include <math.h>

// ---------------- problem dims ----------------
#define BATCH   8
#define SEQ     1024
#define DMODEL  1024
#define DINNER  2048
#define DSTATE  16
#define DCONV   4
#define DTRANK  64
#define DFFN    4096
#define ROWS    (BATCH * SEQ)          // 8192 token rows

typedef __bf16 bf16_t;
typedef __attribute__((ext_vector_type(16))) __bf16 v16bf;
typedef __attribute__((ext_vector_type(8)))  __bf16 v8bf;
typedef __attribute__((ext_vector_type(8)))  float  v8f;

// ---------------------------------------------------------------------------
// fp32 -> bf16 cast (for WMMA inputs)
// ---------------------------------------------------------------------------
__global__ void __launch_bounds__(256)
cast_f32_bf16(const float* __restrict__ in, bf16_t* __restrict__ out, int n) {
    int i = blockIdx.x * blockDim.x + threadIdx.x;
    if (i < n) out[i] = (bf16_t)in[i];
}

// ---------------------------------------------------------------------------
// WMMA fragment loader: row-major [rows, ld] bf16 matrix, 16x32 tile at
// (row0, k0).  ISA 7.12.2 16-bit A layout (B mirrors with N in place of M):
//   lane L: m = L&15, kb = (L>>4)*8
//   halves 0..7  = K [kb    .. kb+7 ]   (16 contiguous bytes)
//   halves 8..15 = K [16+kb .. 16+kb+7] (16 contiguous bytes)
// ---------------------------------------------------------------------------
__device__ __forceinline__ v16bf load_frag16x32(const bf16_t* __restrict__ base,
                                                int ld, int row0, int k0, int lane) {
    int r  = row0 + (lane & 15);
    int kb = (lane >> 4) << 3;
    const bf16_t* p = base + (size_t)r * ld + k0 + kb;
    v8bf lo = *(const v8bf*)(p);
    v8bf hi = *(const v8bf*)(p + 16);
    v16bf f;
#pragma unroll
    for (int i = 0; i < 8; ++i) { f[i] = lo[i]; f[i + 8] = hi[i]; }
    return f;
}

// epilogue modes
#define EPI_F32        0   // outF = acc
#define EPI_DUAL       1   // outF = acc ; outB = bf16(acc)
#define EPI_BIAS_GELU  2   // outB = bf16(gelu(acc + bias))
#define EPI_SOFTPLUS   3   // outF = softplus(acc + bias)
#define EPI_BIAS_ADD   4   // outF = acc + bias + add

// ---------------------------------------------------------------------------
// WMMA GEMM:  out[M,N] = A[M,K] * B[N,K]^T   (A row-major, B row-major over K)
// one wave computes a (16*MT) x (16*NT) register tile:
//   per 32-K step: MT + NT fragment loads feed MT*NT WMMAs.
// MT=2,NT=4 -> 0.75 frag loads/WMMA at ~130 VGPRs: no spills (MT=4 spilled).
// ---------------------------------------------------------------------------
template <int MT, int NT, int EPI>
__global__ void __launch_bounds__(256)
gemm_wmma_bf16(const bf16_t* __restrict__ A, const bf16_t* __restrict__ B,
               float* __restrict__ outF, bf16_t* __restrict__ outB,
               const float* __restrict__ bias, const float* __restrict__ add,
               int M, int N, int K, int lda, int ldb) {
    int wave = blockIdx.x * (blockDim.x >> 5) + (threadIdx.x >> 5);
    int lane = threadIdx.x & 31;
    int groupsN    = N / (16 * NT);
    int totalWaves = (M / (16 * MT)) * groupsN;
    if (wave >= totalWaves) return;           // wave-uniform: EXEC stays all-1s
    int row0 = (wave / groupsN) * 16 * MT;
    int col0 = (wave % groupsN) * 16 * NT;

    v8f acc[MT][NT];
#pragma unroll
    for (int i = 0; i < MT; ++i)
#pragma unroll
        for (int j = 0; j < NT; ++j)
            acc[i][j] = (v8f){0.f, 0.f, 0.f, 0.f, 0.f, 0.f, 0.f, 0.f};

    for (int k0 = 0; k0 < K; k0 += 32) {
        v16bf a[MT];
#pragma unroll
        for (int i = 0; i < MT; ++i)
            a[i] = load_frag16x32(A, lda, row0 + i * 16, k0, lane);
#pragma unroll
        for (int j = 0; j < NT; ++j) {
            v16bf b = load_frag16x32(B, ldb, col0 + j * 16, k0, lane);
#pragma unroll
            for (int i = 0; i < MT; ++i)
                acc[i][j] = __builtin_amdgcn_wmma_f32_16x16x32_bf16(
                    false, a[i], false, b, (short)0, acc[i][j], false, false);
        }
    }

    // C/D layout: lane n = L&15 ; VGPR r -> m = r + (L>=16 ? 8 : 0)
    int nL = lane & 15;
    int mb = (lane >> 4) << 3;
#pragma unroll
    for (int i = 0; i < MT; ++i) {
#pragma unroll
        for (int j = 0; j < NT; ++j) {
            int col = col0 + j * 16 + nL;
#pragma unroll
            for (int r = 0; r < 8; ++r) {
                size_t idx = (size_t)(row0 + i * 16 + mb + r) * N + col;
                float v = acc[i][j][r];
                if (EPI == EPI_F32) {
                    outF[idx] = v;
                } else if (EPI == EPI_DUAL) {
                    outF[idx] = v;
                    outB[idx] = (bf16_t)v;
                } else if (EPI == EPI_BIAS_GELU) {
                    v += bias[col];
                    float g = 0.5f * v * (1.f + erff(v * 0.70710678118654752f));
                    outB[idx] = (bf16_t)g;
                } else if (EPI == EPI_SOFTPLUS) {
                    v += bias[col];
                    outF[idx] = (v > 20.f) ? v : log1pf(expf(v));
                } else { // EPI_BIAS_ADD
                    outF[idx] = v + bias[col] + add[idx];
                }
            }
        }
    }
}

// ---------------------------------------------------------------------------
// causal depthwise conv1d (width 4) + bias + SiLU over xm = xz[:, :, 0:2048]
// ---------------------------------------------------------------------------
__global__ void __launch_bounds__(256)
conv_silu(const float* __restrict__ xz, const float* __restrict__ cw,
          const float* __restrict__ cb,
          float* __restrict__ u, bf16_t* __restrict__ xm_bf) {
    size_t idx = (size_t)blockIdx.x * blockDim.x + threadIdx.x;  // b*l*d_inner
    int ch = (int)(idx % DINNER);
    int t  = (int)((idx / DINNER) % SEQ);
    int b  = (int)(idx / ((size_t)DINNER * SEQ));
    const float* xcol = xz + (size_t)b * SEQ * (2 * DINNER) + ch;
    float s = cb[ch];
#pragma unroll
    for (int k = 0; k < DCONV; ++k) {
        int tt = t - (DCONV - 1) + k;
        if (tt >= 0) s += cw[ch * DCONV + k] * xcol[(size_t)tt * (2 * DINNER)];
    }
    float v = s / (1.f + expf(-s));   // silu
    u[idx]     = v;
    xm_bf[idx] = (bf16_t)v;
}

// ---------------------------------------------------------------------------
// selective scan, one lane per (b, d, n) state element (262144 threads,
// 8192 waves).  Diagonal recurrence: h = exp(dt*A[n])*h + dt*u*B_t[n].
// y_t = sum_n h*C_t[n] reduced across the 16 n-lanes with __shfl_xor
// (wave32 cross-lane, no LDS, no barriers); fused +u*D and *silu(res) gate.
// lane map: threadIdx.x&15 = n, so a wave holds two d channels (16 lanes each)
// ---------------------------------------------------------------------------
__global__ void __launch_bounds__(256)
selective_scan(const float* __restrict__ delta, const float* __restrict__ u,
               const float* __restrict__ xdbl, const float* __restrict__ xz,
               const float* __restrict__ A_log, const float* __restrict__ Dv,
               bf16_t* __restrict__ ygate) {
    int g = (blockIdx.x * blockDim.x + threadIdx.x) >> 4;   // (b,d) channel id
    int n = threadIdx.x & 15;
    int b = g >> 11;                                        // / DINNER
    int d = g & (DINNER - 1);

    float Ad = -expf(A_log[d * DSTATE + n]);
    float Dd = Dv[d];
    float h  = 0.f;

    size_t rowBase = (size_t)b * SEQ;
    for (int t = 0; t < SEQ; ++t) {
        size_t row = rowBase + t;
        float dt = delta[row * DINNER + d];
        float ut = u[row * DINNER + d];
        float Bn = xdbl[row * (DTRANK + 2 * DSTATE) + DTRANK + n];
        float Cn = xdbl[row * (DTRANK + 2 * DSTATE) + DTRANK + DSTATE + n];
        h = expf(dt * Ad) * h + (dt * ut) * Bn;
        float p = h * Cn;
        p += __shfl_xor(p, 1, 32);      // reduce over 16 n-lanes
        p += __shfl_xor(p, 2, 32);
        p += __shfl_xor(p, 4, 32);
        p += __shfl_xor(p, 8, 32);
        if (n == 0) {
            float y   = p + ut * Dd;
            float res = xz[row * (2 * DINNER) + DINNER + d];
            float gsi = res / (1.f + expf(-res));           // silu(res)
            ygate[row * DINNER + d] = (bf16_t)(y * gsi);
        }
    }
}

// ---------------------------------------------------------------------------
// LayerNorm over (seq, d_model) jointly per batch
// ---------------------------------------------------------------------------
__global__ void __launch_bounds__(32)
zero_red(float* __restrict__ red) {
    if (threadIdx.x < 2 * BATCH && blockIdx.x == 0) red[threadIdx.x] = 0.f;
}

__global__ void __launch_bounds__(256)
reduce_stats(const float* __restrict__ c, float* __restrict__ red) {
    const int CHUNK = (SEQ * DMODEL) / 64;         // 16384 elems / block
    int b = blockIdx.x >> 6;
    const float* p = c + (size_t)b * SEQ * DMODEL + (size_t)(blockIdx.x & 63) * CHUNK;
    float s = 0.f, s2 = 0.f;
    for (int i = threadIdx.x; i < CHUNK; i += 256) { float v = p[i]; s += v; s2 += v * v; }
    __shared__ float sh0[256], sh1[256];
    sh0[threadIdx.x] = s; sh1[threadIdx.x] = s2;
    __syncthreads();
    for (int off = 128; off > 0; off >>= 1) {
        if (threadIdx.x < off) {
            sh0[threadIdx.x] += sh0[threadIdx.x + off];
            sh1[threadIdx.x] += sh1[threadIdx.x + off];
        }
        __syncthreads();
    }
    if (threadIdx.x == 0) {
        atomicAdd(&red[2 * b],     sh0[0]);
        atomicAdd(&red[2 * b + 1], sh1[0]);
    }
}

__global__ void __launch_bounds__(256)
layernorm_write(const float* __restrict__ c, const float* __restrict__ red,
                float* __restrict__ out) {
    size_t idx = (size_t)blockIdx.x * blockDim.x + threadIdx.x;
    int b = (int)(idx / ((size_t)SEQ * DMODEL));
    const float invN = 1.f / (float)(SEQ * DMODEL);
    float mean = red[2 * b] * invN;
    float var  = red[2 * b + 1] * invN - mean * mean;
    out[idx] = (c[idx] - mean) * rsqrtf(var + 1e-5f);
}

// ---------------------------------------------------------------------------
// host side
// ---------------------------------------------------------------------------
static inline int gemm_blocks(int M, int N, int MT, int NT) {
    int waves = (M / (16 * MT)) * (N / (16 * NT));
    return (waves + 7) / 8;                        // 8 waves (256 thr) / block
}

extern "C" void kernel_launch(void* const* d_in, const int* in_sizes, int n_in,
                              void* d_out, int out_size, void* d_ws, size_t ws_size,
                              hipStream_t stream) {
    const float* x_in      = (const float*)d_in[0];
    const float* in_proj_w = (const float*)d_in[1];
    const float* conv_w    = (const float*)d_in[2];
    const float* conv_b    = (const float*)d_in[3];
    const float* x_proj_w  = (const float*)d_in[4];
    const float* dt_proj_w = (const float*)d_in[5];
    const float* dt_proj_b = (const float*)d_in[6];
    const float* A_log     = (const float*)d_in[7];
    const float* Dvec      = (const float*)d_in[8];
    const float* out_proj_w= (const float*)d_in[9];
    const float* ffn_w1    = (const float*)d_in[10];
    const float* ffn_b1    = (const float*)d_in[11];
    const float* ffn_w2    = (const float*)d_in[12];
    const float* ffn_b2    = (const float*)d_in[13];
    float* out = (float*)d_out;

    // ---- workspace carve (256B aligned) ----
    size_t off = 0;
    auto carve = [&](size_t bytes) -> void* {
        void* p = (char*)d_ws + off;
        off = (off + bytes + 255) & ~(size_t)255;
        return p;
    };
    bf16_t* xbf      = (bf16_t*)carve((size_t)ROWS * DMODEL * 2);
    bf16_t* w_in_bf  = (bf16_t*)carve((size_t)2 * DINNER * DMODEL * 2);
    bf16_t* w_xp_bf  = (bf16_t*)carve((size_t)(DTRANK + 2 * DSTATE) * DINNER * 2);
    bf16_t* w_dt_bf  = (bf16_t*)carve((size_t)DINNER * DTRANK * 2);
    bf16_t* w_out_bf = (bf16_t*)carve((size_t)DMODEL * DINNER * 2);
    bf16_t* w_f1_bf  = (bf16_t*)carve((size_t)DFFN * DMODEL * 2);
    bf16_t* w_f2_bf  = (bf16_t*)carve((size_t)DMODEL * DFFN * 2);
    float*  xz       = (float*) carve((size_t)ROWS * 2 * DINNER * 4);
    float*  u        = (float*) carve((size_t)ROWS * DINNER * 4);
    bf16_t* xm_bf    = (bf16_t*)carve((size_t)ROWS * DINNER * 2);
    float*  xdbl     = (float*) carve((size_t)ROWS * (DTRANK + 2 * DSTATE) * 4);
    bf16_t* xdbl_bf  = (bf16_t*)carve((size_t)ROWS * (DTRANK + 2 * DSTATE) * 2);
    float*  delta    = (float*) carve((size_t)ROWS * DINNER * 4);
    bf16_t* ygate    = (bf16_t*)carve((size_t)ROWS * DINNER * 2);
    float*  m        = (float*) carve((size_t)ROWS * DMODEL * 4);
    bf16_t* m_bf     = (bf16_t*)carve((size_t)ROWS * DMODEL * 2);
    bf16_t* ffnh     = (bf16_t*)carve((size_t)ROWS * DFFN * 2);
    float*  cbuf     = (float*) carve((size_t)ROWS * DMODEL * 4);
    float*  red      = (float*) carve(2 * BATCH * 4);

    // ---- 1. bf16 casts ----
    auto cast = [&](const float* src, bf16_t* dst, int n) {
        cast_f32_bf16<<<(n + 255) / 256, 256, 0, stream>>>(src, dst, n);
    };
    cast(x_in,       xbf,      ROWS * DMODEL);
    cast(in_proj_w,  w_in_bf,  2 * DINNER * DMODEL);
    cast(x_proj_w,   w_xp_bf,  (DTRANK + 2 * DSTATE) * DINNER);
    cast(dt_proj_w,  w_dt_bf,  DINNER * DTRANK);
    cast(out_proj_w, w_out_bf, DMODEL * DINNER);
    cast(ffn_w1,     w_f1_bf,  DFFN * DMODEL);
    cast(ffn_w2,     w_f2_bf,  DMODEL * DFFN);

    // ---- 2. in_proj: xz[8192,4096] = X * Win^T ----
    gemm_wmma_bf16<2, 4, EPI_F32><<<gemm_blocks(ROWS, 2 * DINNER, 2, 4), 256, 0, stream>>>(
        xbf, w_in_bf, xz, nullptr, nullptr, nullptr,
        ROWS, 2 * DINNER, DMODEL, DMODEL, DMODEL);

    // ---- 3. causal conv + silu -> u (f32) and xm_bf ----
    conv_silu<<<(ROWS * DINNER) / 256, 256, 0, stream>>>(xz, conv_w, conv_b, u, xm_bf);

    // ---- 4. x_proj: xdbl[8192,96] = xm * Wxp^T (dual store f32 + bf16) ----
    gemm_wmma_bf16<2, 2, EPI_DUAL><<<gemm_blocks(ROWS, DTRANK + 2 * DSTATE, 2, 2), 256, 0, stream>>>(
        xm_bf, w_xp_bf, xdbl, xdbl_bf, nullptr, nullptr,
        ROWS, DTRANK + 2 * DSTATE, DINNER, DINNER, DINNER);

    // ---- 5. dt_proj + softplus: delta[8192,2048] ----
    gemm_wmma_bf16<2, 4, EPI_SOFTPLUS><<<gemm_blocks(ROWS, DINNER, 2, 4), 256, 0, stream>>>(
        xdbl_bf, w_dt_bf, delta, nullptr, dt_proj_b, nullptr,
        ROWS, DINNER, DTRANK, DTRANK + 2 * DSTATE, DTRANK);

    // ---- 6. selective scan (+u*D, *silu(res)) -> ygate bf16 ----
    selective_scan<<<(BATCH * DINNER * DSTATE) / 256, 256, 0, stream>>>(
        delta, u, xdbl, xz, A_log, Dvec, ygate);

    // ---- 7. out_proj: m[8192,1024] (dual store) ----
    gemm_wmma_bf16<2, 4, EPI_DUAL><<<gemm_blocks(ROWS, DMODEL, 2, 4), 256, 0, stream>>>(
        ygate, w_out_bf, m, m_bf, nullptr, nullptr,
        ROWS, DMODEL, DINNER, DINNER, DINNER);

    // ---- 8. FFN up + GELU: ffnh[8192,4096] bf16 ----
    gemm_wmma_bf16<2, 4, EPI_BIAS_GELU><<<gemm_blocks(ROWS, DFFN, 2, 4), 256, 0, stream>>>(
        m_bf, w_f1_bf, nullptr, ffnh, ffn_b1, nullptr,
        ROWS, DFFN, DMODEL, DMODEL, DMODEL);

    // ---- 9. FFN down + bias + residual: c = m + f ----
    gemm_wmma_bf16<2, 4, EPI_BIAS_ADD><<<gemm_blocks(ROWS, DMODEL, 2, 4), 256, 0, stream>>>(
        ffnh, w_f2_bf, cbuf, nullptr, ffn_b2, m,
        ROWS, DMODEL, DFFN, DFFN, DFFN);

    // ---- 10. per-batch LayerNorm over (seq, d_model) ----
    zero_red<<<1, 32, 0, stream>>>(red);
    reduce_stats<<<BATCH * 64, 256, 0, stream>>>(cbuf, red);
    layernorm_write<<<(ROWS * DMODEL) / 256, 256, 0, stream>>>(cbuf, red, out);
}
